// ContinuousThoughtMachineGraph_20383914787604
// MI455X (gfx1250) — matrix operand order repaired and checked by reading, not compiled, not run on hardware
//
#include <hip/hip_runtime.h>
#include <math.h>

// ---------------- CDNA5 WMMA types ----------------
typedef __attribute__((ext_vector_type(16))) _Float16 v16h;
typedef __attribute__((ext_vector_type(8)))  _Float16 v8h;
typedef __attribute__((ext_vector_type(8)))  float    v8f;

#define DEV static __device__ __forceinline__

// ---------------- problem dims ----------------
#define B_    128
#define N_    64
#define DIN_  64
#define DI_   128
#define HEADS_ 8
#define DH_   16
#define DM_   512
#define NSA_  64
#define NSO_  64
#define MEM_  25
#define MH_   16
#define ITERS_ 50
#define MAXN_ 64
#define BN_   (B_*N_)   // 8192
#define KSYN_ (DI_+DM_) // 640

DEV float geluf(float x){ return 0.5f*x*(1.0f+erff(x*0.70710678118654752f)); }
DEV float sigmf(float x){ return 1.0f/(1.0f+expf(-x)); }

// 16-bit A fragment (16x32, MxK). lane L holds row M=L&15; K = e + 16*(lane>=16)
DEV int a_kmap(int lane, int e){
  return e + ((lane < 16) ? 0 : 16);
}
// 16-bit B fragment (32x16, KxN). lane L holds col N=L&15; K = e + 16*(lane>=16)
DEV int b_kmap(int lane, int e){ return e + ((lane < 16) ? 0 : 16); }

// A fragment from f16 row-major memory: two contiguous 8-half (16B) chunks.
DEV v16h load_a_frag(const _Float16* rowp, int k0, int lane){
  const int ab = k0 + ((lane < 16) ? 0 : 16);
  v8h lo = *(const v8h*)(rowp + ab);
  v8h hi = *(const v8h*)(rowp + ab + 8);
  return __builtin_shufflevector(lo, hi, 0,1,2,3,4,5,6,7,8,9,10,11,12,13,14,15);
}

// ============================================================
// Generic WMMA GEMM (backbone only, runs once): f32 in/out, f16 ingestion.
// ============================================================
template<int ACT>
__global__ void wmma_gemm_kernel(const float* __restrict__ A, int lda,
                                 const float* __restrict__ W, int ldw,
                                 const float* __restrict__ bias,
                                 float* __restrict__ C, int ldc,
                                 int M, int N, int K)
{
  const int lane = threadIdx.x & 31;
  const int wave = threadIdx.x >> 5;
  const int n0 = blockIdx.x * 16;
  const int m0 = (blockIdx.y * 4 + wave) * 16;
  if (m0 >= M || n0 >= N) return;          // wave-uniform
  const int arow = m0 + (lane & 15);
  const int bcol = n0 + (lane & 15);
  v8f acc = {};
  for (int k0 = 0; k0 < K; k0 += 32) {
    v16h af, bf;
#pragma unroll
    for (int e = 0; e < 16; ++e) {
      int ka = k0 + a_kmap(lane, e);
      af[e] = (_Float16)((ka < K) ? A[(long)arow*lda + ka] : 0.f);
      int kb = k0 + b_kmap(lane, e);
      bf[e] = (_Float16)((kb < K) ? W[(long)kb*ldw + bcol] : 0.f);
    }
    acc = __builtin_amdgcn_wmma_f32_16x16x32_f16(false, af, false, bf,
                                                 (short)0, acc, false, false);
  }
  const int row_base = m0 + ((lane >> 4) << 3);
  const float bs = bias ? bias[bcol] : 0.f;
#pragma unroll
  for (int v = 0; v < 8; ++v) {
    float r = acc[v] + bs;
    if (ACT == 1) r = geluf(r);
    C[(long)(row_base + v)*ldc + bcol] = r;
  }
}

// Row LayerNorm (one wave per row); optional residual-after-LN / GELU-after-LN.
__global__ void ln_kernel(const float* __restrict__ x, const float* __restrict__ g,
                          const float* __restrict__ be, const float* __restrict__ res,
                          float* __restrict__ out, int d, int post_gelu)
{
  const int r = blockIdx.x;
  const int lane = threadIdx.x;  // 32
  const float* xr = x + (long)r*d;
  float s = 0.f, s2 = 0.f;
  for (int i = lane; i < d; i += 32){ float v = xr[i]; s += v; s2 += v*v; }
  for (int off = 16; off; off >>= 1){ s += __shfl_xor(s, off, 32); s2 += __shfl_xor(s2, off, 32); }
  const float mean = s / (float)d;
  const float inv  = rsqrtf(s2/(float)d - mean*mean + 1e-5f);
  for (int i = lane; i < d; i += 32){
    float v = (xr[i]-mean)*inv*g[i] + be[i];
    if (res) v = res[(long)r*d + i] + v;
    else if (post_gelu) v = geluf(v);
    out[(long)r*d + i] = v;
  }
}

// ============================================================
// One-time weight conversions to f16 fragment-native layouts.
// ============================================================
// Wt[n*K + k] = f16(W[k*N + n])  (transposed: B-fragment = one 32B vector load)
__global__ void convert_wt_kernel(const float* __restrict__ W, _Float16* __restrict__ Wt,
                                  int K, int N)
{
  int idx = blockIdx.x*blockDim.x + threadIdx.x;
  if (idx >= N*K) return;
  int n = idx / K, k = idx % K;
  Wt[idx] = (_Float16)W[(long)k*N + n];
}
// NLM bank weights pre-swizzled into exact per-lane B-fragment order (+ nlm2 repack).
__global__ void convert_nlm_kernel(const float* __restrict__ w1, const float* __restrict__ w2,
                                   _Float16* __restrict__ nlmB, float* __restrict__ w2p)
{
  int idx = blockIdx.x*blockDim.x + threadIdx.x;
  if (idx < DM_*2*32){                       // (d, half, lane)
    int lane = idx & 31, half = (idx >> 5) & 1, d = idx >> 6;
    int hcol = lane & 15;
#pragma unroll
    for (int e = 0; e < 16; ++e){
      int m = b_kmap(lane, e);               // K index (MEM padded 25->32)
      float v = (m < MEM_) ? w1[(m*32 + half*16 + hcol)*DM_ + d] : 0.f;
      nlmB[(long)idx*16 + e] = (_Float16)v;
    }
  }
  if (idx < DM_*2*16){                       // w2p[(d*2+j)*16+h] = w2[(h*2+j)*512+d]
    int hh = idx & 15, j = (idx >> 4) & 1, d = idx >> 5;
    w2p[idx] = w2[(hh*2 + j)*DM_ + d];
  }
}

// ============================================================
// PERSISTENT CTM KERNEL: entire 50-step recurrence in one launch.
// grid = 8 blocks (16-row batch tiles), 512 threads = 16 waves (wave r <-> row r).
// Recurrent state lives in LDS; only the 25-deep trace FIFO is in global
// (L2/WGP$-resident, block-private rows).
// ============================================================
__global__ __launch_bounds__(512) void ctm_kernel(
    const float* __restrict__ Kp, const float* __restrict__ Vp,
    const float* __restrict__ q_w, const float* __restrict__ q_b,
    const float* __restrict__ wq,  const float* __restrict__ bq,
    const float* __restrict__ wo,  const float* __restrict__ bo,
    const int* __restrict__ aL, const int* __restrict__ aR,
    const int* __restrict__ oL, const int* __restrict__ oR,
    const float* __restrict__ decayA, const float* __restrict__ decayO,
    const _Float16* __restrict__ synWt, const float* __restrict__ syn_b,
    const float* __restrict__ syn_g, const float* __restrict__ syn_be,
    const _Float16* __restrict__ nlmB, const float* __restrict__ nlm1_b,
    const float* __restrict__ w2p, const float* __restrict__ nlm2_b,
    const float* __restrict__ start_state, const float* __restrict__ start_trace,
    const _Float16* __restrict__ w1t, const float* __restrict__ out_b1,
    const _Float16* __restrict__ w2t, const float* __restrict__ out_b2,
    float* __restrict__ ctrace, float* __restrict__ outp, float* __restrict__ out_sync)
{
  // ---- persistent LDS state ----
  __shared__ __align__(32) float act_s[16][DM_];                   // 32KB f32 master (synch gathers)
  __shared__ __align__(32) _Float16 cat_s[16][KSYN_];              // 20KB f16 [o|act] GEMM A rows
  __shared__ float aA_s[16][NSA_], bA_s[16][NSA_];
  __shared__ float aO_s[16][NSO_], bO_s[16][NSO_];                 // 16KB
  __shared__ __align__(32) _Float16 syncOh_s[16][NSO_];            // 2KB
  __shared__ __align__(32) float h1s[16][16][17];                  // 17KB (per-wave NLM scratch)
  __shared__ __align__(32) union ScratchU {
    struct { float sync[16][NSA_]; float q[16][DI_]; float qh[16][DI_];
             float sc[16][HEADS_][N_]; } attn;                     // 52KB
    float srow[16][2*DM_];                                         // 64KB
    struct { _Float16 h512[16][DM_]; float pred[16][128]; } proj;  // 24KB
  } U;

  const int tid = threadIdx.x, lane = tid & 31, wave = tid >> 5;
  const int r = wave;                       // wave <-> local batch row
  const int b0 = blockIdx.x * 16;

  // ---- init recurrent state (fresh every launch: determinism) ----
  for (int i = tid; i < 16*DM_; i += 512){
    int rr = i >> 9, d = i & 511;
    float v = start_state[d];
    act_s[rr][d] = v;
    cat_s[rr][DI_ + d] = (_Float16)v;
  }
  for (int i = tid; i < 16*NSA_; i += 512){
    int rr = i >> 6, j = i & 63;
    aA_s[rr][j] = 0.f; bA_s[rr][j] = 0.f;
    aO_s[rr][j] = start_state[oL[j]] * start_state[oR[j]];
    bO_s[rr][j] = 1.f;
  }
  for (int i = tid; i < 16*DM_*MEM_; i += 512){
    int p = i % MEM_; int rest = i / MEM_;
    int d = rest & 511, rr = rest >> 9;
    ctrace[((long)((b0+rr)*DM_ + d))*MEM_ + p] = start_trace[d*MEM_ + p];
  }
  // warm the loop weights toward this WGP (global_prefetch_b8 path)
  for (long i = tid*64; i < (long)KSYN_*2*DM_; i += 512*64*8)
    __builtin_prefetch(synWt + i, 0, 3);
  for (long i = tid*64; i < (long)DM_*2*32*16; i += 512*64*8)
    __builtin_prefetch(nlmB + i, 0, 3);
  __threadfence();
  __syncthreads();

  for (int t = 0; t < ITERS_; ++t){
    const int w = t % MEM_;    // circular FIFO write slot (newest)

    // ======== stage 1-3: action synch + q + single-query MHA (wave per row) ========
    for (int jj = lane; jj < NSA_; jj += 32){
      float rd = expf(-decayA[jj]);
      float pp = act_s[r][aL[jj]] * act_s[r][aR[jj]];
      float a  = rd*aA_s[r][jj] + pp;
      float bb = rd*bA_s[r][jj] + 1.f;
      aA_s[r][jj] = a; bA_s[r][jj] = bb;
      U.attn.sync[r][jj] = a * rsqrtf(bb);
    }
    for (int j = lane; j < DI_; j += 32){
      float acc = q_b[j];
      for (int i = 0; i < NSA_; ++i) acc += U.attn.sync[r][i]*q_w[i*DI_+j];
      U.attn.q[r][j] = acc;
    }
    for (int j = lane; j < DI_; j += 32){
      float acc = bq[j];
      for (int i = 0; i < DI_; ++i) acc += U.attn.q[r][i]*wq[i*DI_+j];
      U.attn.qh[r][j] = acc;
    }
    for (int p = lane; p < HEADS_*N_; p += 32){
      int hh = p >> 6, n = p & 63;
      const float* kr = Kp + ((long)((b0+r)*N_+n))*DI_ + hh*DH_;
      float acc = 0.f;
#pragma unroll
      for (int d2 = 0; d2 < DH_; ++d2) acc += U.attn.qh[r][hh*DH_+d2]*kr[d2];
      U.attn.sc[r][hh][n] = acc * 0.25f;            // 1/sqrt(DH)
    }
    if (lane < HEADS_){
      float mx = -1e30f;
      for (int n = 0; n < N_; ++n) mx = fmaxf(mx, U.attn.sc[r][lane][n]);
      float sum = 0.f;
      for (int n = 0; n < N_; ++n){ float e = expf(U.attn.sc[r][lane][n]-mx);
                                    U.attn.sc[r][lane][n] = e; sum += e; }
      float inv = 1.f/sum;
      for (int n = 0; n < N_; ++n) U.attn.sc[r][lane][n] *= inv;
    }
    for (int j = lane; j < DI_; j += 32){           // attn*V (reuse q row)
      int hh = j >> 4, d2 = j & 15;
      const float* vr = Vp + ((long)((b0+r)*N_))*DI_ + hh*DH_ + d2;
      float acc = 0.f;
      for (int n = 0; n < N_; ++n) acc += U.attn.sc[r][hh][n]*vr[n*DI_];
      U.attn.q[r][j] = acc;
    }
    for (int j = lane; j < DI_; j += 32){           // MHA output projection -> f16 A rows
      float acc = bo[j];
      for (int i = 0; i < DI_; ++i) acc += U.attn.q[r][i]*wo[i*DI_+j];
      cat_s[r][j] = (_Float16)acc;
    }
    __syncthreads();

    // ======== stage 4: synapse GEMM  srow[16,1024] = [o|act] @ syn_w ========
    // A fragments: vector ds loads from cat_s. All 4 B fragments preloaded into
    // distinct registers -> one load clause + 4 back-to-back WMMAs per k-step.
    {
      v8f acc0={},acc1={},acc2={},acc3={};
      const int colb = lane & 15;
      const _Float16* arp = &cat_s[lane & 15][0];
      for (int k0 = 0; k0 < KSYN_; k0 += 32){
        v16h af = load_a_frag(arp, k0, lane);
        const _Float16* wp = synWt + k0 + ((lane < 16) ? 0 : 16);
        v16h bf0 = *(const v16h*)(wp + (long)((wave     )*16 + colb)*KSYN_);
        v16h bf1 = *(const v16h*)(wp + (long)((wave + 16)*16 + colb)*KSYN_);
        v16h bf2 = *(const v16h*)(wp + (long)((wave + 32)*16 + colb)*KSYN_);
        v16h bf3 = *(const v16h*)(wp + (long)((wave + 48)*16 + colb)*KSYN_);
        acc0 = __builtin_amdgcn_wmma_f32_16x16x32_f16(false, af, false, bf0,
                                                      (short)0, acc0, false, false);
        acc1 = __builtin_amdgcn_wmma_f32_16x16x32_f16(false, af, false, bf1,
                                                      (short)0, acc1, false, false);
        acc2 = __builtin_amdgcn_wmma_f32_16x16x32_f16(false, af, false, bf2,
                                                      (short)0, acc2, false, false);
        acc3 = __builtin_amdgcn_wmma_f32_16x16x32_f16(false, af, false, bf3,
                                                      (short)0, acc3, false, false);
      }
      const int rb = (lane >> 4) << 3;
#pragma unroll
      for (int i = 0; i < 4; ++i){
        int col = (wave + 16*i)*16 + colb;
        v8f& acc = (i==0?acc0: i==1?acc1: i==2?acc2:acc3);
        float bsv = syn_b[col];
#pragma unroll
        for (int v = 0; v < 8; ++v) U.srow[rb+v][col] = acc[v] + bsv;
      }
    }
    __syncthreads();

    // ======== stage 5: GLU + LayerNorm + trace append (wave per row) ========
    {
      float loc[16]; float sum = 0.f, sum2 = 0.f;
#pragma unroll
      for (int i = 0; i < 16; ++i){
        int d = lane + i*32;
        float g = U.srow[r][d] * sigmf(U.srow[r][d+DM_]);
        loc[i] = g; sum += g; sum2 += g*g;
      }
      for (int off = 16; off; off >>= 1){ sum += __shfl_xor(sum,off,32);
                                          sum2 += __shfl_xor(sum2,off,32); }
      float mean = sum/(float)DM_;
      float inv  = rsqrtf(sum2/(float)DM_ - mean*mean + 1e-5f);
#pragma unroll
      for (int i = 0; i < 16; ++i){
        int d = lane + i*32;
        float v = (loc[i]-mean)*inv*syn_g[d] + syn_be[d];
        U.srow[r][d] = v;                                  // newest state (for NLM m=24)
        ctrace[((long)((b0+r)*DM_+d))*MEM_ + w] = v;       // FIFO write
      }
    }
    __threadfence();
    __syncthreads();

    // ======== stage 6: per-neuron NLM banks, 2 WMMA per neuron ========
    for (int dd = 0; dd < 32; ++dd){
      const int d = wave + dd*16;
      const int arow = lane & 15;
      v16h af;
#pragma unroll
      for (int e = 0; e < 16; ++e){
        int m = a_kmap(lane, e);          // logical memory index (oldest->newest)
        float v = 0.f;
        if (m < MEM_-1){
          int s = w + 1 + m;              // FIFO rotation folded into gather
          int p = (s >= MEM_) ? s - MEM_ : s;
          v = ctrace[((long)((b0+arow)*DM_ + d))*MEM_ + p];
        } else if (m == MEM_-1){
          v = U.srow[arow][d];            // newest slot straight from LDS
        }
        af[e] = (_Float16)v;
      }
      const long nb = (long)(d*2)*32*16;
      v16h blo = *(const v16h*)(nlmB + nb + (long)lane*16);
      v16h bhi = *(const v16h*)(nlmB + nb + 512 + (long)lane*16);
      v8f z = {};
      v8f accA = __builtin_amdgcn_wmma_f32_16x16x32_f16(false, af, false, blo,
                                                        (short)0, z, false, false);
      v8f accB = __builtin_amdgcn_wmma_f32_16x16x32_f16(false, af, false, bhi,
                                                        (short)0, z, false, false);
      const int hcol = lane & 15, rb = (lane >> 4) << 3;
#pragma unroll
      for (int v = 0; v < 8; ++v){
        float a    = accA[v] + nlm1_b[d*32 + hcol];
        float gate = accB[v] + nlm1_b[d*32 + 16 + hcol];
        h1s[wave][rb+v][hcol] = a * sigmf(gate);
      }
      // second bank: K=16, VALU via per-wave LDS + lane swap
      const int j = lane >> 4, bb = lane & 15;
      float acc = nlm2_b[d*2 + j];
#pragma unroll
      for (int hh = 0; hh < MH_; ++hh) acc += h1s[wave][bb][hh] * w2p[(d*2+j)*16 + hh];
      float other = __shfl_xor(acc, 16, 32);
      if (j == 0){
        float nv = acc * sigmf(other);     // new activation
        act_s[bb][d] = nv;
        cat_s[bb][DI_ + d] = (_Float16)nv;
      }
    }
    __syncthreads();

    // ======== stage 7: out-synchronization (wave per row) ========
    for (int jj = lane; jj < NSO_; jj += 32){
      float rd = expf(-decayO[jj]);
      float pp = act_s[r][oL[jj]] * act_s[r][oR[jj]];
      float a  = rd*aO_s[r][jj] + pp;
      float bb = rd*bO_s[r][jj] + 1.f;
      aO_s[r][jj] = a; bO_s[r][jj] = bb;
      float sv = a * rsqrtf(bb);
      syncOh_s[r][jj] = (_Float16)sv;
      if (t == ITERS_-1) out_sync[(b0+r)*NSO_ + jj] = sv;
    }
    __syncthreads();

    // ======== stage 8: output head (2 fused WMMA GEMMs) + certainty ========
    {
      const int colb = lane & 15;
      // GEMM1: [16,64] @ w1t[512][64] -> GELU -> h512 f16 (32 n-tiles / 16 waves)
#pragma unroll
      for (int i = 0; i < 2; ++i){
        const int nt = wave + 16*i;
        v8f acc = {};
        for (int k0 = 0; k0 < NSO_; k0 += 32){
          v16h af = load_a_frag(&syncOh_s[lane & 15][0], k0, lane);
          v16h bf = *(const v16h*)(w1t + (long)(nt*16+colb)*NSO_ + k0 + ((lane<16)?0:16));
          acc = __builtin_amdgcn_wmma_f32_16x16x32_f16(false, af, false, bf,
                                                       (short)0, acc, false, false);
        }
        const int rb = (lane >> 4) << 3;
        float bsv = out_b1[nt*16+colb];
#pragma unroll
        for (int v = 0; v < 8; ++v)
          U.proj.h512[rb+v][nt*16+colb] = (_Float16)geluf(acc[v] + bsv);
      }
      __syncthreads();
      // GEMM2: [16,512] @ w2t[128][512] -> pred f32 (8 n-tiles on waves 0..7)
      if (wave < 8){
        v8f acc = {};
        for (int k0 = 0; k0 < DM_; k0 += 32){
          v16h af = load_a_frag(&U.proj.h512[lane & 15][0], k0, lane);
          v16h bf = *(const v16h*)(w2t + (long)(wave*16+colb)*DM_ + k0 + ((lane<16)?0:16));
          acc = __builtin_amdgcn_wmma_f32_16x16x32_f16(false, af, false, bf,
                                                       (short)0, acc, false, false);
        }
        const int rb = (lane >> 4) << 3;
        float bsv = out_b2[wave*16+colb];
#pragma unroll
        for (int v = 0; v < 8; ++v)
          U.proj.pred[rb+v][wave*16+colb] = acc[v] + bsv;
      }
      __syncthreads();
      // log-softmax over ODIM=2 pairs, normalized entropy, strided scatter
      {
        float nesum = 0.f;
#pragma unroll
        for (int i = 0; i < 2; ++i){
          int n = lane + 32*i;
          float x0 = U.proj.pred[r][2*n], x1 = U.proj.pred[r][2*n+1];
          float m  = fmaxf(x0, x1);
          float l  = m + logf(expf(x0-m)+expf(x1-m));
          float l0 = x0 - l, l1 = x1 - l;
          nesum += -(expf(l0)*l0 + expf(l1)*l1) * 1.4426950408889634f;  // /ln(2)
          outp[((long)((b0+r)*128) + 2*n  )*ITERS_ + t] = x0;
          outp[((long)((b0+r)*128) + 2*n+1)*ITERS_ + t] = x1;
        }
        for (int off = 16; off; off >>= 1) nesum += __shfl_xor(nesum, off, 32);
        if (lane == 0){
          float ne = nesum / (float)MAXN_;
          float* cert = outp + (long)B_*128*ITERS_;
          cert[((long)(b0+r)*2 + 0)*ITERS_ + t] = ne;
          cert[((long)(b0+r)*2 + 1)*ITERS_ + t] = 1.f - ne;
        }
      }
    }
    __syncthreads();   // protect act_s / cat_s / U for next iteration
  }
}

// ============================================================
// Host orchestration
// ============================================================
extern "C" void kernel_launch(void* const* d_in, const int* in_sizes, int n_in,
                              void* d_out, int out_size, void* d_ws, size_t ws_size,
                              hipStream_t stream)
{
  (void)in_sizes; (void)n_in; (void)out_size; (void)ws_size;
  const float* x        = (const float*)d_in[0];
  const float* bb_in_w  = (const float*)d_in[1];
  const float* bb_in_b  = (const float*)d_in[2];
  const float* bb_in_g  = (const float*)d_in[3];
  const float* bb_in_be = (const float*)d_in[4];
  const float* bb_l_w1  = (const float*)d_in[5];
  const float* bb_l_b1  = (const float*)d_in[6];
  const float* bb_l_w2  = (const float*)d_in[7];
  const float* bb_l_b2  = (const float*)d_in[8];
  const float* bb_l_g   = (const float*)d_in[9];
  const float* bb_l_be  = (const float*)d_in[10];
  const float* kv_w     = (const float*)d_in[11];
  const float* kv_b     = (const float*)d_in[12];
  const float* kv_g     = (const float*)d_in[13];
  const float* kv_be    = (const float*)d_in[14];
  const float* q_w      = (const float*)d_in[15];
  const float* q_b      = (const float*)d_in[16];
  const float* att_wq   = (const float*)d_in[17];
  const float* att_bq   = (const float*)d_in[18];
  const float* att_wk   = (const float*)d_in[19];
  const float* att_bk   = (const float*)d_in[20];
  const float* att_wv   = (const float*)d_in[21];
  const float* att_bv   = (const float*)d_in[22];
  const float* att_wo   = (const float*)d_in[23];
  const float* att_bo   = (const float*)d_in[24];
  const float* syn_w    = (const float*)d_in[25];
  const float* syn_b    = (const float*)d_in[26];
  const float* syn_g    = (const float*)d_in[27];
  const float* syn_be   = (const float*)d_in[28];
  const float* nlm1_w   = (const float*)d_in[29];
  const float* nlm1_b   = (const float*)d_in[30];
  const float* nlm2_w   = (const float*)d_in[31];
  const float* nlm2_b   = (const float*)d_in[32];
  const float* start_state = (const float*)d_in[33];
  const float* start_trace = (const float*)d_in[34];
  const float* decayA   = (const float*)d_in[35];
  const float* decayO   = (const float*)d_in[36];
  const float* out_w1   = (const float*)d_in[37];
  const float* out_b1   = (const float*)d_in[38];
  const float* out_w2   = (const float*)d_in[39];
  const float* out_b2   = (const float*)d_in[40];
  const int*   aL       = (const int*)d_in[41];
  const int*   aR       = (const int*)d_in[42];
  const int*   oL       = (const int*)d_in[43];
  const int*   oR       = (const int*)d_in[44];

  // ---- workspace carve-out (floats, 256B-aligned chunks) ----
  float* ws = (float*)d_ws;
  size_t off = 0;
  auto alloc = [&](size_t n){ float* p = ws + off; off += (n + 63) & ~(size_t)63; return p; };
  float* h      = alloc((size_t)BN_*DI_);
  float* t1     = alloc((size_t)BN_*2*DI_);
  float* t2     = alloc((size_t)BN_*DI_);
  float* kv     = alloc((size_t)BN_*DI_);
  float* Kp     = alloc((size_t)BN_*DI_);
  float* Vp     = alloc((size_t)BN_*DI_);
  float* ctrace = alloc((size_t)B_*DM_*MEM_);
  float* w2p    = alloc((size_t)DM_*2*16);
  _Float16* synWt = (_Float16*)alloc((size_t)KSYN_*2*DM_/2);   // [1024][640] f16
  _Float16* w1t   = (_Float16*)alloc((size_t)DM_*NSO_/2);      // [512][64]  f16
  _Float16* w2t   = (_Float16*)alloc((size_t)128*DM_/2);       // [128][512] f16
  _Float16* nlmB  = (_Float16*)alloc((size_t)DM_*2*32*16/2);   // fragment-native

  float* outp     = (float*)d_out;
  float* out_sync = outp + (size_t)B_*128*ITERS_ + (size_t)B_*2*ITERS_;

  const dim3 blk128(128), blk32(32);

  // ---- backbone (once) ----
  wmma_gemm_kernel<0><<<dim3(DI_/16, BN_/64), blk128, 0, stream>>>(
      x, DIN_, bb_in_w, DI_, bb_in_b, t2, DI_, BN_, DI_, DIN_);
  ln_kernel<<<BN_, blk32, 0, stream>>>(t2, bb_in_g, bb_in_be, nullptr, h, DI_, 1);
  for (int lay = 0; lay < 2; ++lay){
    wmma_gemm_kernel<1><<<dim3(2*DI_/16, BN_/64), blk128, 0, stream>>>(
        h, DI_, bb_l_w1 + (size_t)lay*DI_*2*DI_, 2*DI_, bb_l_b1 + lay*2*DI_,
        t1, 2*DI_, BN_, 2*DI_, DI_);
    wmma_gemm_kernel<0><<<dim3(DI_/16, BN_/64), blk128, 0, stream>>>(
        t1, 2*DI_, bb_l_w2 + (size_t)lay*2*DI_*DI_, DI_, bb_l_b2 + lay*DI_,
        t2, DI_, BN_, DI_, 2*DI_);
    ln_kernel<<<BN_, blk32, 0, stream>>>(t2, bb_l_g + lay*DI_, bb_l_be + lay*DI_, h, h, DI_, 0);
  }
  wmma_gemm_kernel<0><<<dim3(DI_/16, BN_/64), blk128, 0, stream>>>(
      h, DI_, kv_w, DI_, kv_b, t2, DI_, BN_, DI_, DI_);
  ln_kernel<<<BN_, blk32, 0, stream>>>(t2, kv_g, kv_be, nullptr, kv, DI_, 0);
  wmma_gemm_kernel<0><<<dim3(DI_/16, BN_/64), blk128, 0, stream>>>(
      kv, DI_, att_wk, DI_, att_bk, Kp, DI_, BN_, DI_, DI_);
  wmma_gemm_kernel<0><<<dim3(DI_/16, BN_/64), blk128, 0, stream>>>(
      kv, DI_, att_wv, DI_, att_bv, Vp, DI_, BN_, DI_, DI_);

  // ---- one-time f16 weight repacks ----
  convert_wt_kernel<<<((2*DM_)*KSYN_+255)/256, 256, 0, stream>>>(syn_w, synWt, KSYN_, 2*DM_);
  convert_wt_kernel<<<(DM_*NSO_+255)/256, 256, 0, stream>>>(out_w1, w1t, NSO_, DM_);
  convert_wt_kernel<<<(128*DM_+255)/256, 256, 0, stream>>>(out_w2, w2t, DM_, 128);
  convert_nlm_kernel<<<(DM_*2*32+255)/256, 256, 0, stream>>>(nlm1_w, nlm2_w, nlmB, w2p);

  // ---- the entire 50-step recurrence: ONE launch ----
  ctm_kernel<<<8, 512, 0, stream>>>(
      Kp, Vp, q_w, q_b, att_wq, att_bq, att_wo, att_bo,
      aL, aR, oL, oR, decayA, decayO,
      synWt, syn_b, syn_g, syn_be,
      nlmB, nlm1_b, w2p, nlm2_b,
      start_state, start_trace,
      w1t, out_b1, w2t, out_b2,
      ctrace, outp, out_sync);
}
